// Attention_472446403161
// MI455X (gfx1250) — matrix operand order repaired and checked by reading, compiled, not run on hardware
//
#include <hip/hip_runtime.h>
#include <hip/hip_bf16.h>

typedef unsigned short u16;
typedef __bf16 bf16_t;
typedef __attribute__((ext_vector_type(16))) __bf16 v16bf;
typedef __attribute__((ext_vector_type(8)))  float  v8f;

// ---------- helpers ----------

__device__ __forceinline__ u16 f2b(float f) {
  unsigned u = __float_as_uint(f);
  u += 0x7fffu + ((u >> 16) & 1u);          // round-to-nearest-even
  return (u16)(u >> 16);
}

__device__ __forceinline__ v8f wmma_bf16(v16bf a, v16bf b, v8f c) {
  return __builtin_amdgcn_wmma_f32_16x16x32_bf16(
      /*neg_a=*/false, a, /*neg_b=*/false, b,
      /*c_mod=*/(short)0, c, /*reuse_a=*/false, /*reuse_b=*/false);
}

// Low 32 bits of a generic pointer to LDS == the LDS byte address (ISA 10.2).
__device__ __forceinline__ unsigned lds_lo32(const void* p) {
  return (unsigned)(unsigned long long)p;
}

// Async global -> LDS copy, 16 bytes per lane; tracked by ASYNCcnt.
__device__ __forceinline__ void async_ld16(unsigned ldsoff, const u16* g) {
  asm volatile("global_load_async_to_lds_b128 %0, %1, off"
               :: "v"(ldsoff), "v"(g) : "memory");
}

__device__ __forceinline__ void wait_async0() {
#if __has_builtin(__builtin_amdgcn_s_wait_asynccnt)
  __builtin_amdgcn_s_wait_asynccnt(0);
#else
  asm volatile("s_wait_asynccnt 0x0" ::: "memory");
#endif
}

union FragU { uint4 q[2]; v16bf v; };

// A-matrix fragment (16x32 bf16, row-major, ld*2 % 16 == 0).
// ISA 7.12.2: lane m=l&15; elems 0..7 = K kh..kh+7, 8..15 = K kh+16..kh+23 (kh=8 hi lanes)
// -> two contiguous 16B runs -> 2x ds_load_b128.
__device__ __forceinline__ v16bf frag_a_vec(const u16* src, int ld) {
  int l  = threadIdx.x & 31;
  int m  = l & 15;
  int kh = (l >> 4) << 3;
  FragU f;
  f.q[0] = *(const uint4*)&src[m * ld + kh];
  f.q[1] = *(const uint4*)&src[m * ld + kh + 16];
  return f.v;
}

// B-matrix fragment (32x16) from K-CONTIGUOUS storage: src[n*ld + koff + k].
// lane n=l&15 = output column; elems 0..7 = K kh..kh+7, 8..15 = K kh+16..kh+23 (kh=8 hi lanes).
__device__ __forceinline__ v16bf frag_k_vec(const u16* src, int ld, int koff) {
  int l  = threadIdx.x & 31;
  int n  = l & 15;
  int kb = (l >> 4) << 3;
  FragU f;
  f.q[0] = *(const uint4*)&src[n * ld + koff + kb];
  f.q[1] = *(const uint4*)&src[n * ld + koff + kb + 16];
  return f.v;
}

// ---------- conversion / layout kernels ----------

__global__ void cvt_bf16_kernel(const float* __restrict__ in, u16* __restrict__ out, int n) {
  int i = blockIdx.x * blockDim.x + threadIdx.x;
  if (i < n) out[i] = f2b(in[i]);
}

// f32 [K x Nn] row-major -> bf16 [Nn x K] (transposed, K-contiguous weights)
__global__ void cvt_t_kernel(const float* __restrict__ in, u16* __restrict__ out,
                             int K, int Nn) {
  int i = blockIdx.x * blockDim.x + threadIdx.x;
  if (i >= K * Nn) return;
  int k = i / Nn, n = i % Nn;
  out[(size_t)n * K + k] = f2b(in[i]);
}

// Extract V from qkv[8192 x 2304] into Vt[b, h, hd, seq] (seq-contiguous).
__global__ void vt_kernel(const u16* __restrict__ qkv, u16* __restrict__ vt) {
  int i = blockIdx.x * blockDim.x + threadIdx.x;   // over 8*1024*768
  if (i >= 8 * 1024 * 768) return;
  int c = i % 768;
  int n = (i / 768) % 1024;
  int b = i / (768 * 1024);
  int h = c >> 6, d = c & 63;
  vt[(((size_t)b * 12 + h) * 64 + d) * 1024 + n] =
      qkv[((size_t)b * 1024 + n) * 2304 + 1536 + c];
}

// ---------- GEMM ----------
// C[M,Nn] = A[M,K] @ B[K,Nn], B supplied TRANSPOSED (Bt[Nn,K], K-contiguous).
// Block: 256 thr = 8 waves; tile 128x64; wave tile 32x32 (4 WMMAs / K-step of 32).
// LDS double-buffered; staging via GLOBAL_LOAD_ASYNC_TO_LDS_B128 (ASYNCcnt).
__global__ void gemm_bf16_kernel(const u16* __restrict__ A, const u16* __restrict__ Bt,
                                 u16* __restrict__ Cb, float* __restrict__ Cf,
                                 const float* __restrict__ bias,
                                 int M, int Nn, int K) {
  __shared__ __align__(16) u16 sA[2][128 * 40];   // 128x32, pad 40
  __shared__ __align__(16) u16 sBt[2][64 * 40];   // 64(N) x 32(K), pad 40

  int ntiles = Nn >> 6;
  int mt = blockIdx.x / ntiles;
  int nt = blockIdx.x % ntiles;
  int mrow0 = mt << 7, ncol0 = nt << 6;

  int wave = threadIdx.x >> 5, lane = threadIdx.x & 31;
  int wm = wave >> 1, wn = wave & 1;

  int cR = threadIdx.x >> 2, cC = (threadIdx.x & 3) << 3;  // 64 rows x 32 cols coop

  const u16* gA0 = &A [(size_t)(mrow0 + cR) * K + cC];
  const u16* gA1 = &A [(size_t)(mrow0 + cR + 64) * K + cC];
  const u16* gB  = &Bt[(size_t)(ncol0 + cR) * K + cC];
  unsigned oA0[2] = { lds_lo32(&sA[0][cR * 40 + cC]),        lds_lo32(&sA[1][cR * 40 + cC]) };
  unsigned oA1[2] = { lds_lo32(&sA[0][(cR + 64) * 40 + cC]), lds_lo32(&sA[1][(cR + 64) * 40 + cC]) };
  unsigned oB [2] = { lds_lo32(&sBt[0][cR * 40 + cC]),       lds_lo32(&sBt[1][cR * 40 + cC]) };

  auto stage = [&](int bufi, int k0) {
    async_ld16(oA0[bufi], gA0 + k0);
    async_ld16(oA1[bufi], gA1 + k0);
    async_ld16(oB [bufi], gB  + k0);
  };

  stage(0, 0);
  wait_async0();
  __syncthreads();

  v8f acc00 = {}, acc01 = {}, acc10 = {}, acc11 = {};
  int buf = 0;
  for (int k0 = 0; k0 < K; k0 += 32) {
    int nb = buf ^ 1;
    if (k0 + 32 < K) {
      stage(nb, k0 + 32);                                 // overlap with WMMAs below
      if (k0 + 64 < K) __builtin_prefetch(gA0 + k0 + 64, 0, 1);
    }

    const u16* pA = &sA[buf][0];
    const u16* pB = &sBt[buf][0];
    v16bf a0 = frag_a_vec(&pA[(wm * 32) * 40], 40);
    v16bf a1 = frag_a_vec(&pA[(wm * 32 + 16) * 40], 40);
    v16bf b0 = frag_k_vec(&pB[(wn * 32) * 40], 40, 0);
    v16bf b1 = frag_k_vec(&pB[(wn * 32 + 16) * 40], 40, 0);
    acc00 = wmma_bf16(a0, b0, acc00);
    acc01 = wmma_bf16(a0, b1, acc01);
    acc10 = wmma_bf16(a1, b0, acc10);
    acc11 = wmma_bf16(a1, b1, acc11);

    wait_async0();       // own async loads into next buffer done
    __syncthreads();     // everyone done reading cur + everyone's loads visible
    buf = nb;
  }

  int n  = lane & 15;
  int mb = (lane >> 4) << 3;                 // C layout: hi lanes hold M=r+8
  int row0 = mrow0 + wm * 32 + mb;
  int col0 = ncol0 + wn * 32;
  const v8f* accs[4] = { &acc00, &acc01, &acc10, &acc11 };
#pragma unroll
  for (int t = 0; t < 4; ++t) {
    int rr = row0 + (t >> 1) * 16;
    int cc = col0 + (t & 1) * 16 + n;
    const v8f& a = *accs[t];
    if (Cb) {
#pragma unroll
      for (int r = 0; r < 8; ++r)
        Cb[(size_t)(rr + r) * Nn + cc] = f2b(a[r]);
    } else {
      float bv = bias ? bias[cc] : 0.f;
#pragma unroll
      for (int r = 0; r < 8; ++r)
        Cf[(size_t)(rr + r) * Nn + cc] = a[r] + bv;
    }
  }
}

// ---------- Flash attention ----------
// qkv bf16 [8192 x 2304] (q|k|v, 12 heads x 64), vt bf16 [b,h,64,1024].
// grid = 8 * 12 * (1024/128); block = 256 thr = 8 waves, wave = 16 Q rows.
// K/V tiles double-buffered with async LDS loads.
__global__ void flash_attn_kernel(const u16* __restrict__ qkv,
                                  const u16* __restrict__ vt,
                                  u16* __restrict__ attnb) {
  __shared__ __align__(16) u16 sK[2][32 * 72];     // 32 keys x 64 hd
  __shared__ __align__(16) u16 sVt[2][64 * 40];    // 64 hd x 32 keys (key-contig)
  __shared__ __align__(16) u16 sP[8][16 * 40];     // per-wave P tile 16x32

  int blk = blockIdx.x;
  int rb  = blk & 7;
  int h   = (blk >> 3) % 12;
  int b   = blk / 96;

  int wave = threadIdx.x >> 5, lane = threadIdx.x & 31;
  int qrow0 = rb * 128 + wave * 16;

  const u16* qbase  = qkv + ((size_t)b * 1024) * 2304 + h * 64;
  const u16* kbase  = qbase + 768;
  const u16* vtbase = vt + (((size_t)b * 12 + h) * 64) * 1024;

  // Q fragments (K-dim 64 -> two 16x32 A-frags), b128 global loads.
  v16bf qa0 = frag_a_vec(qbase + (size_t)qrow0 * 2304, 2304);
  v16bf qa1 = frag_a_vec(qbase + (size_t)qrow0 * 2304 + 32, 2304);

  float mrow[8], lrow[8];
  v8f o0 = {}, o1 = {}, o2 = {}, o3 = {};
#pragma unroll
  for (int r = 0; r < 8; ++r) { mrow[r] = -3.0e38f; lrow[r] = 0.f; }

  int kR = threadIdx.x >> 3, kC = (threadIdx.x & 7) << 3;  // 32x64 coop
  int vR = threadIdx.x >> 2, vC = (threadIdx.x & 3) << 3;  // 64x32 coop
  const u16* gK = &kbase [(size_t)kR * 2304 + kC];
  const u16* gV = &vtbase[(size_t)vR * 1024 + vC];
  unsigned oK[2] = { lds_lo32(&sK[0][kR * 72 + kC]),  lds_lo32(&sK[1][kR * 72 + kC]) };
  unsigned oV[2] = { lds_lo32(&sVt[0][vR * 40 + vC]), lds_lo32(&sVt[1][vR * 40 + vC]) };

  auto stage = [&](int bufi, int j) {
    async_ld16(oK[bufi], gK + (size_t)j * 2304);
    async_ld16(oV[bufi], gV + j);
  };

  const float scale = 0.125f;                              // 64^-0.5

  stage(0, 0);
  wait_async0();
  __syncthreads();

  int buf = 0;
  for (int j = 0; j < 1024; j += 32) {
    int nb = buf ^ 1;
    if (j + 32 < 1024) stage(nb, j + 32);                  // overlap with compute

    const u16* pK = &sK[buf][0];
    const u16* pV = &sVt[buf][0];

    // S = Q @ K^T : 16 rows x 32 keys (2 tiles; K-dim 64 -> 2 WMMAs each)
    v8f s0 = {}, s1 = {};
    s0 = wmma_bf16(qa0, frag_k_vec(&pK[0],       72, 0),  s0);
    s0 = wmma_bf16(qa1, frag_k_vec(&pK[0],       72, 32), s0);
    s1 = wmma_bf16(qa0, frag_k_vec(&pK[16 * 72], 72, 0),  s1);
    s1 = wmma_bf16(qa1, frag_k_vec(&pK[16 * 72], 72, 32), s1);

    // Online softmax (rows live in 16-lane halves per the C-matrix layout).
    float corr[8];
#pragma unroll
    for (int r = 0; r < 8; ++r) {
      float a0 = s0[r] * scale, a1 = s1[r] * scale;
      float rm = fmaxf(a0, a1);
      rm = fmaxf(rm, __shfl_xor(rm, 1, 32));
      rm = fmaxf(rm, __shfl_xor(rm, 2, 32));
      rm = fmaxf(rm, __shfl_xor(rm, 4, 32));
      rm = fmaxf(rm, __shfl_xor(rm, 8, 32));
      float mnew = fmaxf(mrow[r], rm);
      float p0 = __expf(a0 - mnew);
      float p1 = __expf(a1 - mnew);
      float c  = __expf(mrow[r] - mnew);
      float rs = p0 + p1;
      rs += __shfl_xor(rs, 1, 32);
      rs += __shfl_xor(rs, 2, 32);
      rs += __shfl_xor(rs, 4, 32);
      rs += __shfl_xor(rs, 8, 32);
      lrow[r] = lrow[r] * c + rs;
      mrow[r] = mnew;
      corr[r] = c;
      s0[r] = p0; s1[r] = p1;
    }
#pragma unroll
    for (int r = 0; r < 8; ++r) {
      o0[r] *= corr[r]; o1[r] *= corr[r]; o2[r] *= corr[r]; o3[r] *= corr[r];
    }

    // C-layout P -> row-major LDS (wave-private; same-wave LDS in-order) -> A-frag
    int nn = lane & 15, mb = (lane >> 4) << 3;
    u16* P = &sP[wave][0];
#pragma unroll
    for (int r = 0; r < 8; ++r) {
      P[(mb + r) * 40 + nn]      = f2b(s0[r]);
      P[(mb + r) * 40 + 16 + nn] = f2b(s1[r]);
    }
    v16bf pa = frag_a_vec(P, 40);

    // O(16x64) += P(16x32) @ V(32x64)
    o0 = wmma_bf16(pa, frag_k_vec(&pV[ 0 * 40], 40, 0), o0);
    o1 = wmma_bf16(pa, frag_k_vec(&pV[16 * 40], 40, 0), o1);
    o2 = wmma_bf16(pa, frag_k_vec(&pV[32 * 40], 40, 0), o2);
    o3 = wmma_bf16(pa, frag_k_vec(&pV[48 * 40], 40, 0), o3);

    wait_async0();
    __syncthreads();
    buf = nb;
  }

  int nn = lane & 15, mb = (lane >> 4) << 3;
  size_t obase = ((size_t)b * 1024 + qrow0 + mb) * 768 + h * 64 + nn;
#pragma unroll
  for (int r = 0; r < 8; ++r) {
    float inv = (lrow[r] > 0.f) ? 1.f / lrow[r] : 0.f;
    size_t o = obase + (size_t)r * 768;
    attnb[o]      = f2b(o0[r] * inv);
    attnb[o + 16] = f2b(o1[r] * inv);
    attnb[o + 32] = f2b(o2[r] * inv);
    attnb[o + 48] = f2b(o3[r] * inv);
  }
}

// ---------- launcher ----------

extern "C" void kernel_launch(void* const* d_in, const int* in_sizes, int n_in,
                              void* d_out, int out_size, void* d_ws, size_t ws_size,
                              hipStream_t stream) {
  const float* x      = (const float*)d_in[0];   // [8,1024,768]
  const float* w_qkv  = (const float*)d_in[1];   // [768,2304]
  const float* w_proj = (const float*)d_in[2];   // [768,768]
  const float* b_proj = (const float*)d_in[3];   // [768]

  const int M  = 8 * 1024;      // 8192 rows
  const int C  = 768;
  const int C3 = 2304;

  u16* xb    = (u16*)d_ws;                         // 8192x768
  u16* wqkbT = xb    + (size_t)M * C;              // 2304x768 (transposed)
  u16* wpbT  = wqkbT + (size_t)C * C3;             // 768x768  (transposed)
  u16* qkv   = wpbT  + (size_t)C * C;              // 8192x2304
  u16* vtb   = qkv   + (size_t)M * C3;             // b,h,64,1024
  u16* attn  = vtb   + (size_t)8 * 12 * 64 * 1024; // 8192x768

  int n1 = M * C;
  cvt_bf16_kernel<<<(n1 + 255) / 256, 256, 0, stream>>>(x, xb, n1);
  int n2 = C * C3;
  cvt_t_kernel<<<(n2 + 255) / 256, 256, 0, stream>>>(w_qkv, wqkbT, C, C3);
  int n3 = C * C;
  cvt_t_kernel<<<(n3 + 255) / 256, 256, 0, stream>>>(w_proj, wpbT, C, C);

  // QKV GEMM: [8192,2304] = xb @ w_qkv (bf16 out), B pre-transposed
  gemm_bf16_kernel<<<(M / 128) * (C3 / 64), 256, 0, stream>>>(
      xb, wqkbT, qkv, nullptr, nullptr, M, C3, C);

  // V -> Vt[b,h,hd,seq]
  int n4 = 8 * 1024 * 768;
  vt_kernel<<<(n4 + 255) / 256, 256, 0, stream>>>(qkv, vtb);

  // Flash attention
  flash_attn_kernel<<<768, 256, 0, stream>>>(qkv, vtb, attn);

  // Projection: out = attn @ w_proj + b_proj (f32 out), B pre-transposed
  gemm_bf16_kernel<<<(M / 128) * (C / 64), 256, 0, stream>>>(
      attn, wpbT, nullptr, (float*)d_out, b_proj, M, C, C);
}
